// ScaledDotProductAttention_15685220565605
// MI455X (gfx1250) — matrix operand order repaired and checked by reading, not compiled, not run on hardware
//
#include <hip/hip_runtime.h>

// Problem constants (B=2, H=16, S=2048, D=64)
#define BH      32
#define SEQ     2048
#define DIM     64
#define BLOCK_M 64      // queries per workgroup (16 per wave)
#define BLOCK_N 64      // keys per inner tile
#define NWAVES  4       // 128 threads, wave32

typedef __bf16 bf16;
typedef bf16  v16bf __attribute__((ext_vector_type(16)));
typedef float v8f   __attribute__((ext_vector_type(8)));

__launch_bounds__(128, 2)
__global__ void fa_fwd_kernel(const float* __restrict__ qg,
                              const float* __restrict__ kg,
                              const float* __restrict__ vg,
                              const float* __restrict__ biasg,
                              float* __restrict__ og)
{
    // LDS staging: K row-major (bf16), V transposed (bf16), per-wave P buffer.
    // +8 bf16 pad -> 36-dword row stride: 16-lane b128 row-strided reads hit
    // all 64 banks exactly once (conflict-free).
    __shared__ bf16 sK [BLOCK_N][DIM + 8];        // [key][d]
    __shared__ bf16 sVt[DIM][BLOCK_N + 8];        // [d][key]
    __shared__ bf16 sP [NWAVES][16][BLOCK_N + 8]; // [wave][row][key]

    const int bh   = blockIdx.y;          // batch*head
    const int tid  = threadIdx.x;
    const int wave = tid >> 5;
    const int lane = tid & 31;
    const int lh   = lane >> 4;           // 16-lane half (0/1)
    const int ln   = lane & 15;

    const float* qb = qg + (size_t)bh * SEQ * DIM;
    const float* kb = kg + (size_t)bh * SEQ * DIM;
    const float* vb = vg + (size_t)bh * SEQ * DIM;
    const float* bb = biasg + (size_t)bh * SEQ;
    float*       ob = og + (size_t)bh * SEQ * DIM;

    const int q0 = blockIdx.x * BLOCK_M + wave * 16;   // first query row of this wave

    // ---- Q A-fragments (16x32 bf16, pre-scaled by 1/sqrt(64)) ----
    // A 16x32 layout: lane holds row=ln; elem e -> K = 16*lh+e
    v16bf aq[2];
    {
        const float* qrow = qb + (size_t)(q0 + ln) * DIM;
        #pragma unroll
        for (int c = 0; c < 2; ++c) {
            #pragma unroll
            for (int e = 0; e < 8; ++e) {
                aq[c][e]     = (bf16)(0.125f * qrow[c * 32 + 16 * lh + e]);
                aq[c][e + 8] = (bf16)(0.125f * qrow[c * 32 + 16 * lh + 8 + e]);
            }
        }
    }

    // Per-query bias for the 8 rows this lane covers in C-layout (M = 8*lh + i)
    float brow[8];
    #pragma unroll
    for (int i = 0; i < 8; ++i) brow[i] = bb[q0 + 8 * lh + i];

    // Online-softmax state and output accumulators (fp32)
    float mrow[8], lrow[8];
    #pragma unroll
    for (int i = 0; i < 8; ++i) { mrow[i] = -1e30f; lrow[i] = 0.0f; }

    v8f acc[4];
    #pragma unroll
    for (int j = 0; j < 4; ++j)
        #pragma unroll
        for (int i = 0; i < 8; ++i) acc[j][i] = 0.0f;

    for (int n0 = 0; n0 < SEQ; n0 += BLOCK_N) {
        __syncthreads();
        // ---- cooperative stage: K tile (row-major) + V tile (transposed) ----
        // 64x64 f32, float4-vectorized: 1024 vec4 / 128 threads = 8 each.
        for (int idx = tid; idx < BLOCK_N * DIM / 4; idx += 128) {
            const int kk = idx >> 4;             // key in tile (16 float4 per row)
            const int d4 = (idx & 15) << 2;      // dim base
            const float4 kf = *(const float4*)(kb + (size_t)(n0 + kk) * DIM + d4);
            const float4 vf = *(const float4*)(vb + (size_t)(n0 + kk) * DIM + d4);
            sK[kk][d4 + 0] = (bf16)kf.x;
            sK[kk][d4 + 1] = (bf16)kf.y;
            sK[kk][d4 + 2] = (bf16)kf.z;
            sK[kk][d4 + 3] = (bf16)kf.w;
            sVt[d4 + 0][kk] = (bf16)vf.x;
            sVt[d4 + 1][kk] = (bf16)vf.y;
            sVt[d4 + 2][kk] = (bf16)vf.z;
            sVt[d4 + 3][kk] = (bf16)vf.w;
            if (n0 + BLOCK_N < SEQ) {
                __builtin_prefetch(kb + (size_t)(n0 + BLOCK_N + kk) * DIM + d4, 0, 0);
                __builtin_prefetch(vb + (size_t)(n0 + BLOCK_N + kk) * DIM + d4, 0, 0);
            }
        }
        __syncthreads();

        // ---- S = Q K^T : four 16x16 score tiles, each K=64 via 2 WMMA ----
        v8f sc[4];
        #pragma unroll
        for (int t = 0; t < 4; ++t) {
            v8f c0;
            #pragma unroll
            for (int i = 0; i < 8; ++i) c0[i] = 0.0f;
            #pragma unroll
            for (int c = 0; c < 2; ++c) {
                // B 32x16 layout: lane col n=ln; elem e -> Kdim = 16*lh + e
                v16bf bk;
                #pragma unroll
                for (int e = 0; e < 16; ++e)
                    bk[e] = sK[t * 16 + ln][c * 32 + 16 * lh + e];
                c0 = __builtin_amdgcn_wmma_f32_16x16x32_bf16(
                        false, aq[c], false, bk, (short)0, c0, false, false);
            }
            sc[t] = c0;
        }

        // ---- bias + online softmax (row = 8*lh+i, cols across 16-lane half) ----
        float mnew[8], rs[8];
        #pragma unroll
        for (int i = 0; i < 8; ++i) {
            float mx = -1e30f;
            #pragma unroll
            for (int t = 0; t < 4; ++t) {
                const float s = sc[t][i] + brow[i];
                sc[t][i] = s;
                mx = fmaxf(mx, s);
            }
            #pragma unroll
            for (int off = 1; off < 16; off <<= 1)
                mx = fmaxf(mx, __shfl_xor(mx, off, 32));
            mnew[i] = fmaxf(mrow[i], mx);
            rs[i]   = __expf(mrow[i] - mnew[i]);
            mrow[i] = mnew[i];
        }

        // exp + P store (C-layout -> per-wave LDS) + row-sum
        #pragma unroll
        for (int i = 0; i < 8; ++i) {
            float ps = 0.0f;
            #pragma unroll
            for (int t = 0; t < 4; ++t) {
                const float p = __expf(sc[t][i] - mnew[i]);
                sP[wave][8 * lh + i][t * 16 + ln] = (bf16)p;
                ps += p;
            }
            #pragma unroll
            for (int off = 1; off < 16; off <<= 1)
                ps += __shfl_xor(ps, off, 32);
            lrow[i] = lrow[i] * rs[i] + ps;
        }

        // ---- re-read P in A-layout (two 16x32 chunks) ----
        v16bf ap[2];
        #pragma unroll
        for (int c = 0; c < 2; ++c) {
            #pragma unroll
            for (int e = 0; e < 8; ++e) {
                ap[c][e]     = sP[wave][ln][c * 32 + 16 * lh + e];
                ap[c][e + 8] = sP[wave][ln][c * 32 + 16 * lh + 8 + e];
            }
        }

        // ---- O = diag(rs)*O + P V : 4 d-chunks of 16, K=64 via 2 WMMA ----
        #pragma unroll
        for (int j = 0; j < 4; ++j) {
            #pragma unroll
            for (int i = 0; i < 8; ++i) acc[j][i] *= rs[i];
            #pragma unroll
            for (int c = 0; c < 2; ++c) {
                // B 32x16: lane col n=ln -> d = 16j+ln ; elem e -> key = c*32+16*lh+e
                v16bf bv;
                #pragma unroll
                for (int e = 0; e < 16; ++e)
                    bv[e] = sVt[j * 16 + ln][c * 32 + 16 * lh + e];
                acc[j] = __builtin_amdgcn_wmma_f32_16x16x32_bf16(
                            false, ap[c], false, bv, (short)0, acc[j], false, false);
            }
        }
    }

    // ---- normalize and store (row M = 8*lh+i, col d = 16j+ln) ----
    #pragma unroll
    for (int j = 0; j < 4; ++j) {
        #pragma unroll
        for (int i = 0; i < 8; ++i) {
            const float o = acc[j][i] / lrow[i];
            ob[(size_t)(q0 + 8 * lh + i) * DIM + j * 16 + ln] = o;
        }
    }
}

extern "C" void kernel_launch(void* const* d_in, const int* in_sizes, int n_in,
                              void* d_out, int out_size, void* d_ws, size_t ws_size,
                              hipStream_t stream) {
    const float* q    = (const float*)d_in[0];
    const float* k    = (const float*)d_in[1];
    const float* v    = (const float*)d_in[2];
    const float* bias = (const float*)d_in[3];
    float* out = (float*)d_out;

    dim3 grid(SEQ / BLOCK_M, BH);   // (32, 32)
    dim3 block(128);                // 4 waves of 32
    fa_fwd_kernel<<<grid, block, 0, stream>>>(q, k, v, bias, out);
}
